// GMPEncoder_83047487635785
// MI455X (gfx1250) — compile-verified, matrix operand order checked by reading, not run on hardware
//
#include <hip/hip_runtime.h>
#include <hip/hip_bf16.h>
#include <stdint.h>

// ---------------------------------------------------------------------------
// Bidirectional jump-LSTM encoder for MI455X (gfx1250, wave32, WMMA).
// bf16 WMMA (f32 accumulate) for all GEMMs; recurrent scan = ONE workgroup per
// direction, h carried in double-buffered LDS bf16 (1 barrier/step), cell
// state + running maxes pinned in WMMA accumulator registers for all 512
// steps. Streaming buffers (Gx, f32 outputs) use nontemporal cache hints so
// L2 stays hot for W_hh (re-read every step) and the bf16 outputs copy.
// ---------------------------------------------------------------------------

typedef __attribute__((ext_vector_type(16))) __bf16 v16bf;
typedef __attribute__((ext_vector_type(8)))  __bf16 v8bf;
typedef __attribute__((ext_vector_type(8)))  float  v8f;

#define BB   64      // batch
#define TT   512     // time
#define DD   256     // embedding dim
#define HH   256     // hidden
#define G4   1024    // 4*H (gates)
#define H2   512     // 2*H (bidir concat)

__device__ __forceinline__ __bf16 f2bf(float f) {
  // round-to-nearest-even f32 -> bf16 via bit ops
  unsigned u = __builtin_bit_cast(unsigned, f);
  unsigned r = (u + 0x7FFFu + ((u >> 16) & 1u)) >> 16;
  unsigned short s = (unsigned short)r;
  return __builtin_bit_cast(__bf16, s);
}

__device__ __forceinline__ float sigm(float x) { return 1.0f / (1.0f + __expf(-x)); }

union Frag16 { v16bf v; v8bf h[2]; };

// 16x32 bf16 A/B fragment per CDNA5 ISA §7.12.2:
// lanes 0-15 hold K=[k0..k0+7] and [k0+16..k0+23]; lanes 16-31 mirrored halves.
// Two 16-byte loads per lane (global_load_b128 / ds_load_b128 after inlining).
__device__ __forceinline__ v16bf load_frag(const __bf16* base, int row, int ld,
                                           int k0, int lane) {
  int half = lane >> 4;
  const __bf16* p = base + (long)row * ld + k0 + half * 8;
  Frag16 f;
  f.h[0] = *(const v8bf*)p;
  f.h[1] = *(const v8bf*)(p + 16);
  return f.v;
}

__device__ __forceinline__ v8f wmma_bf16(v16bf a, v16bf b, v8f c) {
  // D(f32 16x16) = A(bf16 16x32) * B(bf16 32x16) + C
  return __builtin_amdgcn_wmma_f32_16x16x32_bf16(false, a, false, b, (short)0, c,
                                                 false, false);
}

// ---------------------------------------------------------------------------
// Kernel 1: f32 -> bf16 weight cast (one-shot, tiny)
// ---------------------------------------------------------------------------
__global__ void k_cast(const float* __restrict__ s, __bf16* __restrict__ d, int n) {
  int i = blockIdx.x * blockDim.x + threadIdx.x;
  if (i < n) d[i] = f2bf(s[i]);
}

// ---------------------------------------------------------------------------
// Kernel 2: embedding gather (f32 -> bf16) + pad_mask output
// ---------------------------------------------------------------------------
__global__ void k_embed(const int* __restrict__ seqs, const float* __restrict__ emb,
                        __bf16* __restrict__ xbf, float* __restrict__ pmask) {
  int bt = blockIdx.x;
  int d = threadIdx.x;
  int tok = seqs[bt];
  __builtin_prefetch(emb + (long)tok * DD, 0, 0);   // global_prefetch_b8
  xbf[(long)bt * DD + d] = f2bf(emb[(long)tok * DD + d]);
  if (d == 0) pmask[bt] = (tok > 0) ? 1.0f : -1.0f;
}

// ---------------------------------------------------------------------------
// Kernel 3: input GEMM  Gx[dir] = x(_rev) @ W_ih^T + b_ih   (M=32768,N=1024,K=256)
// Time reversal for the backward direction folded into the A-row mapping so
// the scan reads Gx sequentially. Gx stores are nontemporal (streaming,
// 256 MB total > 192 MB L2).
// ---------------------------------------------------------------------------
__global__ __launch_bounds__(128) void k_in_gemm(const __bf16* __restrict__ xbf,
                                                 const __bf16* __restrict__ wih,
                                                 const float* __restrict__ bih,
                                                 float* __restrict__ gx, int dir) {
  int wave = threadIdx.x >> 5, lane = threadIdx.x & 31;
  int half = lane >> 4, l15 = lane & 15;
  long tile = (long)blockIdx.x * 4 + wave;       // 0 .. 2048*64-1
  int nt = (int)(tile & 63);
  int mt = (int)(tile >> 6);
  int row = mt * 16 + l15;                       // scan-order row = b*T + t
  int b = row >> 9, t = row & 511;
  int src_row = (b << 9) | (dir ? (511 - t) : t);
  int wrow = nt * 16 + l15;
  v8f acc = {};
  for (int k0 = 0; k0 < DD; k0 += 32) {
    v16bf a = load_frag(xbf, src_row, DD, k0, lane);
    v16bf w = load_frag(wih, wrow, DD, k0, lane);
    acc = wmma_bf16(a, w, acc);
  }
  int colg = nt * 16 + l15;                      // C-frag: N = lane&15
  float bias = bih[colg];
  #pragma unroll
  for (int r = 0; r < 8; r++) {
    int m = mt * 16 + r + 8 * half;              // C-frag: M = r + 8*(lane>=16)
    __builtin_nontemporal_store(acc[r] + bias, &gx[(long)m * G4 + colg]);
  }
}

// ---------------------------------------------------------------------------
// Kernel 4: recurrent jump-LSTM scan. grid=2 (dir), block=1024 (32 waves).
// Double-buffered LDS h-state => exactly ONE s_barrier per time step.
// Per step: Gx gate pre-activations prefetched into registers, then
// G = Gx + h @ W_hh^T + b_hh via WMMA (A-frags from LDS, B-frags from L2),
// gates applied on accumulator fragments, carry written straight to the
// opposite LDS buffer. Each wave owns 2 fixed (batch-tile, hidden-tile)
// blocks => c-state and h/c maxes live in registers for all 512 steps.
// ---------------------------------------------------------------------------
__global__ __launch_bounds__(1024, 1) void k_scan(
    const float* __restrict__ gx_f, const float* __restrict__ gx_b,
    const __bf16* __restrict__ whh_f, const __bf16* __restrict__ whh_b,
    const float* __restrict__ bhh_f, const float* __restrict__ bhh_b,
    const int* __restrict__ seqs, const int* __restrict__ jump,
    float* __restrict__ out, __bf16* __restrict__ outbf,
    float* __restrict__ hm_out, float* __restrict__ cm_out) {
  __shared__ __bf16 hbuf[2][BB * HH];   // 64 KB double-buffered hidden state
  __shared__ float smask[2][BB];
  __shared__ float sjmp[2][BB];

  const int dir = blockIdx.x;
  const int tid = threadIdx.x;
  const int lane = tid & 31;
  const int wave = tid >> 5;
  const int half = lane >> 4;
  const int l15 = lane & 15;

  const float*  gx  = dir ? gx_b  : gx_f;
  const __bf16* whh = dir ? whh_b : whh_f;
  const float*  bhh = dir ? bhh_b : bhh_f;

  for (int i = tid; i < BB * HH; i += 1024) hbuf[0][i] = f2bf(0.0f);

  v8f creg[2]; v8f hmax[2]; v8f cmax[2];
  #pragma unroll
  for (int q = 0; q < 2; q++)
    #pragma unroll
    for (int r = 0; r < 8; r++) {
      creg[q][r] = 0.0f; hmax[q][r] = -3.0e38f; cmax[q][r] = -3.0e38f;
    }

  // stage masks for t=0
  if (tid < BB) {
    int idx = tid * TT + (dir ? (TT - 1) : 0);
    smask[0][tid] = (seqs[idx] > 0) ? 1.0f : 0.0f;
    sjmp[0][tid] = (float)jump[idx];
  }
  __syncthreads();

  for (int t = 0; t < TT; t++) {
    const int p = t & 1;
    const int tsrc = dir ? (TT - 1 - t) : t;

    // stage masks for t+1 into the opposite buffer (no conflict with readers)
    if (t + 1 < TT && tid < BB) {
      int idx = tid * TT + (dir ? (TT - 2 - t) : (t + 1));
      smask[p ^ 1][tid] = (seqs[idx] > 0) ? 1.0f : 0.0f;
      sjmp[p ^ 1][tid] = (float)jump[idx];
    }

    #pragma unroll
    for (int q = 0; q < 2; q++) {
      const int blk = wave * 2 + q;      // 0..63
      const int mt = blk >> 4;           // batch tile 0..3
      const int ht = blk & 15;           // hidden tile 0..15
      const int hcol = ht * 16 + l15;

      // prefetch Gx gate pre-activations (overlaps the WMMA k-loop below)
      float gxi[8], gxf[8], gxg[8], gxo[8];
      #pragma unroll
      for (int r = 0; r < 8; r++) {
        const int b = mt * 16 + r + 8 * half;
        const long gbase = ((long)b * TT + t) * G4 + hcol;
        gxi[r] = __builtin_nontemporal_load(&gx[gbase]);
        gxf[r] = __builtin_nontemporal_load(&gx[gbase + HH]);
        gxg[r] = __builtin_nontemporal_load(&gx[gbase + 2 * HH]);
        gxo[r] = __builtin_nontemporal_load(&gx[gbase + 3 * HH]);
      }

      v8f acc0 = {}, acc1 = {}, acc2 = {}, acc3 = {};
      for (int k0 = 0; k0 < HH; k0 += 32) {
        v16bf a = load_frag(hbuf[p], mt * 16 + l15, HH, k0, lane); // ds_load_b128
        v16bf b0 = load_frag(whh, 0 * HH + ht * 16 + l15, HH, k0, lane);
        v16bf b1 = load_frag(whh, 1 * HH + ht * 16 + l15, HH, k0, lane);
        v16bf b2 = load_frag(whh, 2 * HH + ht * 16 + l15, HH, k0, lane);
        v16bf b3 = load_frag(whh, 3 * HH + ht * 16 + l15, HH, k0, lane);
        acc0 = wmma_bf16(a, b0, acc0);   // i
        acc1 = wmma_bf16(a, b1, acc1);   // f
        acc2 = wmma_bf16(a, b2, acc2);   // g
        acc3 = wmma_bf16(a, b3, acc3);   // o
      }

      #pragma unroll
      for (int r = 0; r < 8; r++) {
        const int b = mt * 16 + r + 8 * half;
        float gi = acc0[r] + gxi[r] + bhh[hcol];
        float gf = acc1[r] + gxf[r] + bhh[HH + hcol];
        float gg = acc2[r] + gxg[r] + bhh[2 * HH + hcol];
        float go = acc3[r] + gxo[r] + bhh[3 * HH + hcol];
        float cn = sigm(gf) * creg[q][r] + sigm(gi) * tanhf(gg);
        float hn = sigm(go) * tanhf(cn);
        hmax[q][r] = fmaxf(hmax[q][r], hn);
        cmax[q][r] = fmaxf(cmax[q][r], cn);
        float ho = hn * smask[p][b];
        long oidx = ((long)b * TT + tsrc) * H2 + dir * HH + hcol;
        __builtin_nontemporal_store(ho, &out[oidx]);  // f32 outputs (result #1)
        outbf[oidx] = f2bf(ho);          // bf16 copy, reused by projection GEMM
        float j = sjmp[p][b];
        creg[q][r] = cn * j;             // jump-gated carry (registers)
        hbuf[p ^ 1][b * HH + hcol] = f2bf(hn * j);  // carry -> opposite buffer
      }
    }
    __syncthreads();   // single barrier: publishes hbuf[p^1] + masks for t+1
  }

  // epilogue: h_max / c_max (results #3/#4); fragments map to fixed (b,hcol)
  #pragma unroll
  for (int q = 0; q < 2; q++) {
    const int blk = wave * 2 + q;
    const int mt = blk >> 4, ht = blk & 15;
    const int hcol = ht * 16 + l15;
    #pragma unroll
    for (int r = 0; r < 8; r++) {
      const int b = mt * 16 + r + 8 * half;
      long idx = (long)dir * BB * HH + (long)b * HH + hcol;
      hm_out[idx] = hmax[q][r];
      cm_out[idx] = cmax[q][r];
    }
  }
}

// ---------------------------------------------------------------------------
// Kernel 5: encoder_feature = outputs @ W_h^T   (M=32768, N=512, K=512)
// ---------------------------------------------------------------------------
__global__ __launch_bounds__(128) void k_out_gemm(const __bf16* __restrict__ abf,
                                                  const __bf16* __restrict__ whbf,
                                                  float* __restrict__ ef) {
  int wave = threadIdx.x >> 5, lane = threadIdx.x & 31;
  int half = lane >> 4, l15 = lane & 15;
  long tile = (long)blockIdx.x * 4 + wave;       // 0 .. 2048*32-1
  int nt = (int)(tile & 31);
  int mt = (int)(tile >> 5);
  int arow = mt * 16 + l15;
  int wrow = nt * 16 + l15;
  __builtin_prefetch(whbf + (long)wrow * H2, 0, 1);
  v8f acc = {};
  for (int k0 = 0; k0 < H2; k0 += 32) {
    v16bf a = load_frag(abf, arow, H2, k0, lane);
    v16bf w = load_frag(whbf, wrow, H2, k0, lane);
    acc = wmma_bf16(a, w, acc);
  }
  int col = nt * 16 + l15;
  #pragma unroll
  for (int r = 0; r < 8; r++) {
    int m = mt * 16 + r + 8 * half;
    __builtin_nontemporal_store(acc[r], &ef[(long)m * H2 + col]);
  }
}

// ---------------------------------------------------------------------------
// Launcher
// ---------------------------------------------------------------------------
extern "C" void kernel_launch(void* const* d_in, const int* in_sizes, int n_in,
                              void* d_out, int out_size, void* d_ws, size_t ws_size,
                              hipStream_t stream) {
  const int*   seqs = (const int*)d_in[0];
  const int*   jmp  = (const int*)d_in[1];
  const float* emb  = (const float*)d_in[2];
  const float* wihf = (const float*)d_in[3];
  const float* whhf = (const float*)d_in[4];
  const float* bihf = (const float*)d_in[5];
  const float* bhhf = (const float*)d_in[6];
  const float* wihb = (const float*)d_in[7];
  const float* whhb = (const float*)d_in[8];
  const float* bihb = (const float*)d_in[9];
  const float* bhhb = (const float*)d_in[10];
  const float* wh   = (const float*)d_in[11];

  float* out = (float*)d_out;                      // tuple flat, return order:
  float* outputs = out;                            // (B,T,2H)
  float* efeat = outputs + (long)BB * TT * H2;     // (B*T,2H)
  float* hm = efeat + (long)BB * TT * H2;          // (2,B,H)
  float* cm = hm + 2L * BB * HH;                   // (2,B,H)
  float* pmask = cm + 2L * BB * HH;                // (B,T)

  char* ws = (char*)d_ws;
  size_t o = 0;
  auto take = [&](size_t bytes) -> char* {
    char* p = ws + o;
    o += (bytes + 255) & ~(size_t)255;
    return p;
  };
  __bf16* xbf   = (__bf16*)take((size_t)BB * TT * DD * 2);   //  16 MB
  float*  gxF   = (float*) take((size_t)BB * TT * G4 * 4);   // 128 MB
  float*  gxB   = (float*) take((size_t)BB * TT * G4 * 4);   // 128 MB
  __bf16* wihfB = (__bf16*)take((size_t)G4 * DD * 2);
  __bf16* wihbB = (__bf16*)take((size_t)G4 * DD * 2);
  __bf16* whhfB = (__bf16*)take((size_t)G4 * HH * 2);
  __bf16* whhbB = (__bf16*)take((size_t)G4 * HH * 2);
  __bf16* whB   = (__bf16*)take((size_t)H2 * H2 * 2);
  __bf16* outbf = (__bf16*)take((size_t)BB * TT * H2 * 2);   //  32 MB
  (void)ws_size; (void)in_sizes; (void)n_in; (void)out_size;

  k_cast<<<(G4 * DD + 255) / 256, 256, 0, stream>>>(wihf, wihfB, G4 * DD);
  k_cast<<<(G4 * DD + 255) / 256, 256, 0, stream>>>(wihb, wihbB, G4 * DD);
  k_cast<<<(G4 * HH + 255) / 256, 256, 0, stream>>>(whhf, whhfB, G4 * HH);
  k_cast<<<(G4 * HH + 255) / 256, 256, 0, stream>>>(whhb, whhbB, G4 * HH);
  k_cast<<<(H2 * H2 + 255) / 256, 256, 0, stream>>>(wh, whB, H2 * H2);

  k_embed<<<BB * TT, DD, 0, stream>>>(seqs, emb, xbf, pmask);

  k_in_gemm<<<(2048 * 64) / 4, 128, 0, stream>>>(xbf, wihfB, bihf, gxF, 0);
  k_in_gemm<<<(2048 * 64) / 4, 128, 0, stream>>>(xbf, wihbB, bihb, gxB, 1);

  k_scan<<<2, 1024, 0, stream>>>(gxF, gxB, whhfB, whhbB, bhhf, bhhb, seqs, jmp,
                                 outputs, outbf, hm, cm);

  k_out_gemm<<<(2048 * 32) / 4, 128, 0, stream>>>(outbf, whB, efeat);
}
